// EncoderBlock_10307921510973
// MI455X (gfx1250) — matrix operand order repaired and checked by reading, not compiled, not run on hardware
//
#include <hip/hip_runtime.h>

// ---------------------------------------------------------------------------
// CDNA5 WMMA types (wave32)
// ---------------------------------------------------------------------------
typedef __attribute__((ext_vector_type(16))) __bf16 v16bf;
typedef __attribute__((ext_vector_type(8)))  __bf16 v8bf;
typedef __attribute__((ext_vector_type(8)))  float  v8f;

__device__ inline v16bf join16(v8bf a, v8bf b) {
  return __builtin_shufflevector(a, b, 0,1,2,3,4,5,6,7,8,9,10,11,12,13,14,15);
}
__device__ inline v8f wmma_bf(v16bf a, v16bf b, v8f c) {
  return __builtin_amdgcn_wmma_f32_16x16x32_bf16(false, a, false, b,
                                                 (short)0, c, false, false);
}

// ---------------------------------------------------------------------------
// CDNA5 async global->LDS copy (ASYNCcnt path) + waits
// ---------------------------------------------------------------------------
__device__ inline unsigned lds_off(const void* p) {
  return (unsigned)(uintptr_t)p;  // LDS byte address = addr[31:0]
}
__device__ inline void async_ld_b128(unsigned lds, unsigned gbyteoff,
                                     unsigned long long gbase) {
  asm volatile("global_load_async_to_lds_b128 %0, %1, %2"
               :: "v"(lds), "v"(gbyteoff), "s"(gbase) : "memory");
}
__device__ inline void wait_async() {
  asm volatile("s_wait_asynccnt 0x0" ::: "memory");
}
__device__ inline void wait_ds() {
  asm volatile("s_wait_dscnt 0x0" ::: "memory");
}

// ---------------------------------------------------------------------------
// Tiled cast + transpose: in f32 [K][N] -> out bf16 [N][K]  (weights)
// ---------------------------------------------------------------------------
__global__ __launch_bounds__(256)
void cast_transpose(const float* __restrict__ in, __bf16* __restrict__ out,
                    int K, int N) {
  __shared__ __bf16 t[64][72];
  const int k0 = blockIdx.y * 64, n0 = blockIdx.x * 64;
  const int r = threadIdx.x >> 2, c4 = (threadIdx.x & 3) * 16;
  #pragma unroll
  for (int j = 0; j < 16; j++)
    t[c4 + j][r] = (__bf16)in[(size_t)(k0 + r) * N + n0 + c4 + j];
  __syncthreads();
  #pragma unroll
  for (int j = 0; j < 16; j++)
    out[(size_t)(n0 + r) * K + k0 + c4 + j] = t[r][c4 + j];
}

// ---------------------------------------------------------------------------
// LayerNorm (scalar g/b, unbiased std) -> bf16. One block per row.
// ---------------------------------------------------------------------------
__global__ __launch_bounds__(256)
void layernorm_kernel(const float* __restrict__ x, const float* __restrict__ g,
                      const float* __restrict__ b, __bf16* __restrict__ out,
                      int D) {
  const int row = blockIdx.x;
  const float* xr = x + (size_t)row * D;
  float s = 0.f, ss = 0.f;
  for (int i = threadIdx.x; i < D; i += 256) {
    float v = xr[i];
    s += v; ss += v * v;
  }
  #pragma unroll
  for (int off = 1; off < 32; off <<= 1) {
    s  += __shfl_xor(s,  off, 32);
    ss += __shfl_xor(ss, off, 32);
  }
  __shared__ float rs[2][8];
  const int wv = threadIdx.x >> 5;
  if ((threadIdx.x & 31) == 0) { rs[0][wv] = s; rs[1][wv] = ss; }
  __syncthreads();
  float S = 0.f, SS = 0.f;
  #pragma unroll
  for (int i = 0; i < 8; i++) { S += rs[0][i]; SS += rs[1][i]; }
  const float mean = S / D;
  const float var  = (SS - S * mean) / (D - 1);
  const float inv  = 1.0f / (sqrtf(var) + 1e-6f);
  const float gg = g[0], bb = b[0];
  __bf16* orow = out + (size_t)row * D;
  for (int i = threadIdx.x; i < D; i += 256)
    orow[i] = (__bf16)(gg * (xr[i] - mean) * inv + bb);
}

// ---------------------------------------------------------------------------
// WMMA GEMM: C[M,N] = oscale*(A[M,K] @ Bt[N,K]^T + bias), then relu/residual.
// Block tile 128x128x64, double-buffered LDS filled by async b128 copies.
// 8 waves, each 32x64 (2x4 fragments) -> 16 WMMAs per K-tile.
// VSC: scatter output as per-head-transposed V: [B*H][64][S].
// ---------------------------------------------------------------------------
template <bool BIAS, bool RELU, bool RES, bool OUTF, bool OUTB, bool VSC>
__global__ __launch_bounds__(256)
void gemm_tn(const __bf16* __restrict__ A, const __bf16* __restrict__ Bt,
             const float* __restrict__ bias, const float* __restrict__ residual,
             float* __restrict__ outF, __bf16* __restrict__ outB,
             __bf16* __restrict__ outVT, int M, int N, int K, int sshift,
             float oscale) {
  __shared__ __bf16 As[2][128][64];
  __shared__ __bf16 Bs[2][128][64];
  const int tid  = threadIdx.x;
  const int lane = tid & 31, wv = tid >> 5;
  const int r = lane & 15, hi = lane >> 4;
  const int m0 = blockIdx.y * 128, n0 = blockIdx.x * 128;
  const int wm = (wv & 3) * 32, wn = (wv >> 2) * 64;

  const unsigned long long aBase = (unsigned long long)A;
  const unsigned long long bBase = (unsigned long long)Bt;

  auto stage = [&](int buf, int k0) {
    #pragma unroll
    for (int c = 0; c < 4; c++) {
      const int cf = tid * 4 + c;
      const int row = cf >> 3, cc = (cf & 7) * 8;
      async_ld_b128(lds_off(&As[buf][row][cc]),
                    (unsigned)(((size_t)(m0 + row) * K + k0 + cc) * 2), aBase);
      async_ld_b128(lds_off(&Bs[buf][row][cc]),
                    (unsigned)(((size_t)(n0 + row) * K + k0 + cc) * 2), bBase);
    }
  };

  v8f acc[2][4] = {};

  stage(0, 0);
  wait_async();
  __syncthreads();

  const int T = K >> 6;
  for (int t = 0; t < T; t++) {
    const int nb = t & 1;
    if (t + 1 < T) stage(nb ^ 1, (t + 1) << 6);

    #pragma unroll
    for (int ks = 0; ks < 64; ks += 32) {
      v16bf af[2], bfv[4];
      #pragma unroll
      for (int mi = 0; mi < 2; mi++) {
        const __bf16* ap = &As[nb][wm + mi * 16 + r][ks];
        af[mi] = join16(*(const v8bf*)(ap + hi * 8),
                        *(const v8bf*)(ap + 16 + hi * 8));
      }
      #pragma unroll
      for (int ni = 0; ni < 4; ni++) {
        const __bf16* bp = &Bs[nb][wn + ni * 16 + r][ks];
        bfv[ni] = join16(*(const v8bf*)(bp + hi * 8),
                         *(const v8bf*)(bp + 16 + hi * 8));
      }
      #pragma unroll
      for (int mi = 0; mi < 2; mi++)
        #pragma unroll
        for (int ni = 0; ni < 4; ni++)
          acc[mi][ni] = wmma_bf(af[mi], bfv[ni], acc[mi][ni]);
    }
    if (t + 1 < T) {
      wait_async();
      __syncthreads();
    }
  }

  // Branch-free epilogue (all options compile-time)
  const int Smask = (1 << sshift) - 1;
  #pragma unroll
  for (int mi = 0; mi < 2; mi++)
    #pragma unroll
    for (int ni = 0; ni < 4; ni++) {
      const int nn = n0 + wn + ni * 16 + r;
      float bv = 0.f;
      if (BIAS) bv = bias[nn];
      #pragma unroll
      for (int i = 0; i < 8; i++) {
        const int mm = m0 + wm + mi * 16 + i + 8 * hi;
        float v = acc[mi][ni][i];
        if (BIAS) v += bv;
        if (RELU) v = fmaxf(v, 0.f);
        v *= oscale;
        const size_t idx = (size_t)mm * N + nn;
        if (RES)  v += residual[idx];
        if (OUTF) outF[idx] = v;
        if (OUTB) outB[idx] = (__bf16)v;
        if (VSC) {
          const int bb = mm >> sshift, s = mm & Smask;
          const int h = nn >> 6, d = nn & 63;
          outVT[(((size_t)bb * 16 + h) * 64 + d) << sshift | s] = (__bf16)v;
        }
      }
    }
}

// ---------------------------------------------------------------------------
// Flash attention. Q bf16 [B*S,1024] (head at col h*64, pre-scaled by 1/8);
// K bf16 [B*S,1024]; VT bf16 per-head transposed [B*H][64][S].
// Grid (S/128, B*H), 256 threads; 64 keys per iteration (16 WMMAs/iter),
// double-buffered async K/V tiles, online softmax.
// ---------------------------------------------------------------------------
__global__ __launch_bounds__(256)
void attn_kernel(const __bf16* __restrict__ Q, const __bf16* __restrict__ Kb,
                 const __bf16* __restrict__ VT, const int* __restrict__ mask,
                 __bf16* __restrict__ ctx, int S) {
  __shared__ __bf16 Ks[2][64][64];    // [key][d]
  __shared__ __bf16 Vt[2][64][64];    // [d][key]
  __shared__ __bf16 Ps[8][16][64];    // per-wave P tile [row][key]
  const int tid = threadIdx.x, lane = tid & 31, wv = tid >> 5;
  const int r = lane & 15, hi = lane >> 4;
  const int bh = blockIdx.y, b = bh >> 4, h = bh & 15;
  const int q0 = blockIdx.x * 128 + wv * 16;

  const unsigned long long kBase = (unsigned long long)Kb;
  const unsigned long long vBase = (unsigned long long)VT;

  auto stage = [&](int buf, int kk) {
    #pragma unroll
    for (int c = 0; c < 2; c++) {
      const int cf = tid * 2 + c;
      const int row = cf >> 3, cc = (cf & 7) * 8;  // row 0..63, cc 0..56
      async_ld_b128(lds_off(&Ks[buf][row][cc]),
          (unsigned)(((size_t)(b * S + kk + row) * 1024 + h * 64 + cc) * 2),
          kBase);
      async_ld_b128(lds_off(&Vt[buf][row][cc]),
          (unsigned)(((size_t)(bh * 64 + row) * S + kk + cc) * 2), vBase);
    }
  };

  // Preload Q fragments (d 0..31, 32..63); Q already scaled by 1/sqrt(dk)
  const __bf16* qp = Q + (size_t)(b * S + q0 + r) * 1024 + h * 64;
  const v16bf qa0 = join16(*(const v8bf*)(qp + hi * 8),
                           *(const v8bf*)(qp + 16 + hi * 8));
  const v16bf qa1 = join16(*(const v8bf*)(qp + 32 + hi * 8),
                           *(const v8bf*)(qp + 48 + hi * 8));

  float mrow[8], lrow[8];
  #pragma unroll
  for (int i = 0; i < 8; i++) { mrow[i] = -1e30f; lrow[i] = 0.f; }
  v8f cacc[4] = {};

  stage(0, 0);
  wait_async();
  __syncthreads();

  for (int kk = 0; kk < S; kk += 64) {
    const int nb = (kk >> 6) & 1;
    if (kk + 64 < S) stage(nb ^ 1, kk + 64);

    // Scores S[16q x 64k]: 4 key sub-tiles x 2 d-steps
    v8f sc[4];
    #pragma unroll
    for (int t2 = 0; t2 < 4; t2++) {
      const __bf16* kp = &Ks[nb][16 * t2 + r][0];
      v8f a = {};
      a = wmma_bf(qa0, join16(*(const v8bf*)(kp + hi * 8),
                              *(const v8bf*)(kp + 16 + hi * 8)), a);
      a = wmma_bf(qa1, join16(*(const v8bf*)(kp + 32 + hi * 8),
                              *(const v8bf*)(kp + 48 + hi * 8)), a);
      const float mk = mask[(size_t)b * S + kk + 16 * t2 + r] ? 0.f : -1e9f;
      #pragma unroll
      for (int i = 0; i < 8; i++) a[i] += mk;
      sc[t2] = a;
    }

    // Online softmax; row (i + 8*hi) lives in the 16 lanes of one half-wave
    #pragma unroll
    for (int i = 0; i < 8; i++) {
      float pm = fmaxf(fmaxf(sc[0][i], sc[1][i]), fmaxf(sc[2][i], sc[3][i]));
      #pragma unroll
      for (int off = 1; off < 16; off <<= 1)
        pm = fmaxf(pm, __shfl_xor(pm, off, 16));
      const float mn = fmaxf(mrow[i], pm);
      const float alpha = __expf(mrow[i] - mn);
      mrow[i] = mn;
      float ps = 0.f;
      #pragma unroll
      for (int t2 = 0; t2 < 4; t2++) {
        const float e = __expf(sc[t2][i] - mn);
        sc[t2][i] = e;
        ps += e;
      }
      #pragma unroll
      for (int off = 1; off < 16; off <<= 1) ps += __shfl_xor(ps, off, 16);
      lrow[i] = lrow[i] * alpha + ps;
      #pragma unroll
      for (int c = 0; c < 4; c++) cacc[c][i] *= alpha;
    }

    // D-layout -> A-layout transpose of P through per-wave LDS
    #pragma unroll
    for (int t2 = 0; t2 < 4; t2++)
      #pragma unroll
      for (int i = 0; i < 8; i++)
        Ps[wv][i + 8 * hi][16 * t2 + r] = (__bf16)sc[t2][i];
    wait_ds();
    const __bf16* pp = &Ps[wv][r][0];
    const v16bf pa0 = join16(*(const v8bf*)(pp + hi * 8),
                             *(const v8bf*)(pp + 16 + hi * 8));
    const v16bf pa1 = join16(*(const v8bf*)(pp + 32 + hi * 8),
                             *(const v8bf*)(pp + 48 + hi * 8));

    // ctx += P(16x64) @ V(64x64): 4 d-chunks x 2 k-steps
    #pragma unroll
    for (int c = 0; c < 4; c++) {
      const __bf16* vp = &Vt[nb][16 * c + r][0];
      cacc[c] = wmma_bf(pa0, join16(*(const v8bf*)(vp + hi * 8),
                                    *(const v8bf*)(vp + 16 + hi * 8)),
                        cacc[c]);
      cacc[c] = wmma_bf(pa1, join16(*(const v8bf*)(vp + 32 + hi * 8),
                                    *(const v8bf*)(vp + 48 + hi * 8)),
                        cacc[c]);
    }
    if (kk + 64 < S) {
      wait_async();
      __syncthreads();
    }
  }

  // Normalize, write ctx bf16 [B*S,1024] head at col h*64
  #pragma unroll
  for (int i = 0; i < 8; i++) {
    const float inv = 1.0f / lrow[i];
    const int row = q0 + i + 8 * hi;
    __bf16* op = ctx + (size_t)(b * S + row) * 1024 + h * 64 + r;
    #pragma unroll
    for (int c = 0; c < 4; c++) op[16 * c] = (__bf16)(cacc[c][i] * inv);
  }
}

// ---------------------------------------------------------------------------
// Host orchestration
// ---------------------------------------------------------------------------
extern "C" void kernel_launch(void* const* d_in, const int* in_sizes, int n_in,
                              void* d_out, int out_size, void* d_ws,
                              size_t ws_size, hipStream_t stream) {
  const int B = 4, S = 2048, D = 1024, DFF = 4096;
  const int M = B * S;        // 8192
  const int SSH = 11;         // log2(S)

  const float* x    = (const float*)d_in[0];
  const int*   mask = (const int*)d_in[1];
  const float* wq   = (const float*)d_in[2];
  const float* bq   = (const float*)d_in[3];
  const float* wk   = (const float*)d_in[4];
  const float* bk   = (const float*)d_in[5];
  const float* wv   = (const float*)d_in[6];
  const float* bv   = (const float*)d_in[7];
  const float* wo   = (const float*)d_in[8];
  const float* bo   = (const float*)d_in[9];
  const float* w1   = (const float*)d_in[10];
  const float* b1   = (const float*)d_in[11];
  const float* w2   = (const float*)d_in[12];
  const float* b2   = (const float*)d_in[13];
  const float* ln1g = (const float*)d_in[14];
  const float* ln1b = (const float*)d_in[15];
  const float* ln2g = (const float*)d_in[16];
  const float* ln2b = (const float*)d_in[17];
  float* out = (float*)d_out;

  char* ws = (char*)d_ws;
  size_t off = 0;
  auto alloc = [&](size_t bytes) -> void* {
    void* p = ws + off;
    off += (bytes + 255) & ~(size_t)255;
    return p;
  };
  __bf16* h1b  = (__bf16*)alloc((size_t)M * D * 2);
  __bf16* wqt  = (__bf16*)alloc((size_t)D * D * 2);     // [N][K]
  __bf16* wkt  = (__bf16*)alloc((size_t)D * D * 2);
  __bf16* wvt  = (__bf16*)alloc((size_t)D * D * 2);
  __bf16* wot  = (__bf16*)alloc((size_t)D * D * 2);
  __bf16* w1t  = (__bf16*)alloc((size_t)D * DFF * 2);   // [4096][1024]
  __bf16* w2t  = (__bf16*)alloc((size_t)DFF * D * 2);   // [1024][4096]
  __bf16* qb   = (__bf16*)alloc((size_t)M * D * 2);
  __bf16* kb   = (__bf16*)alloc((size_t)M * D * 2);
  __bf16* vtb  = (__bf16*)alloc((size_t)M * D * 2);     // [B*H][64][S]
  __bf16* ctxb = (__bf16*)alloc((size_t)M * D * 2);
  float*  x2   = (float*)alloc((size_t)M * D * 4);
  __bf16* h2b  = (__bf16*)alloc((size_t)M * D * 2);
  __bf16* ff1b = (__bf16*)alloc((size_t)M * DFF * 2);

  // 1) weight cast + transpose: w[K][N] -> wt[N][K] bf16
  dim3 gT1(D / 64, D / 64);
  cast_transpose<<<gT1, 256, 0, stream>>>(wq, wqt, D, D);
  cast_transpose<<<gT1, 256, 0, stream>>>(wk, wkt, D, D);
  cast_transpose<<<gT1, 256, 0, stream>>>(wv, wvt, D, D);
  cast_transpose<<<gT1, 256, 0, stream>>>(wo, wot, D, D);
  dim3 gT2(DFF / 64, D / 64);
  cast_transpose<<<gT2, 256, 0, stream>>>(w1, w1t, D, DFF);
  dim3 gT3(D / 64, DFF / 64);
  cast_transpose<<<gT3, 256, 0, stream>>>(w2, w2t, DFF, D);

  // 2) LN1
  layernorm_kernel<<<M, 256, 0, stream>>>(x, ln1g, ln1b, h1b, D);

  // 3) Q/K/V projections (Q pre-scaled by 1/sqrt(dk)=0.125)
  dim3 gQ(D / 128, M / 128);
  gemm_tn<true,false,false,false,true,false><<<gQ, 256, 0, stream>>>(
      h1b, wqt, bq, nullptr, nullptr, qb, nullptr, M, D, D, SSH, 0.125f);
  gemm_tn<true,false,false,false,true,false><<<gQ, 256, 0, stream>>>(
      h1b, wkt, bk, nullptr, nullptr, kb, nullptr, M, D, D, SSH, 1.0f);
  gemm_tn<true,false,false,false,false,true><<<gQ, 256, 0, stream>>>(
      h1b, wvt, bv, nullptr, nullptr, nullptr, vtb, M, D, D, SSH, 1.0f);

  // 4) flash attention
  dim3 gA(S / 128, B * 16);
  attn_kernel<<<gA, 256, 0, stream>>>(qb, kb, vtb, mask, ctxb, S);

  // 5) output projection + residual -> x2 (f32)
  gemm_tn<true,false,true,true,false,false><<<gQ, 256, 0, stream>>>(
      ctxb, wot, bo, x, x2, nullptr, nullptr, M, D, D, SSH, 1.0f);

  // 6) LN2
  layernorm_kernel<<<M, 256, 0, stream>>>(x2, ln2g, ln2b, h2b, D);

  // 7) FFN1 + ReLU (bf16)
  dim3 gF1(DFF / 128, M / 128);
  gemm_tn<true,true,false,false,true,false><<<gF1, 256, 0, stream>>>(
      h2b, w1t, b1, nullptr, nullptr, ff1b, nullptr, M, DFF, D, SSH, 1.0f);

  // 8) FFN2 + residual -> d_out (f32)
  dim3 gF2(D / 128, M / 128);
  gemm_tn<true,false,true,true,false,false><<<gF2, 256, 0, stream>>>(
      ff1b, w2t, b2, x2, out, nullptr, nullptr, M, D, DFF, SSH, 1.0f);
}
